// WReN_48962627174550
// MI455X (gfx1250) — matrix operand order repaired and checked by reading, not compiled, not run on hardware
//
#include <hip/hip_runtime.h>

// WReN relation network for MI455X (gfx1250), wave32 + WMMA f16 (f32 accum).
//
// Shapes: B=128, Q=32, N_WAY=2, N_SHOT=6, D=256. NUM_EMB=7, 42 ordered pairs.
// Restructured: per-embedding W1 projection (PA = emb@W1[:256], PB = emb@W1[256:]),
// shot-shot pair sums hoisted out of the query loop, pad rows corrected analytically.

typedef __attribute__((ext_vector_type(16))) _Float16 v16h;
typedef __attribute__((ext_vector_type(8)))  float    v8f;

#define WMMA_F16(a, b, c) \
  __builtin_amdgcn_wmma_f32_16x16x32_f16(false, (a), false, (b), (short)0, (c), false, false)

// 16-bit A-matrix 16x32 per-lane layout (CDNA5 ISA 7.12.2):
// lanes 0-15: V0..V3 -> K 0..7, V4..V7 -> K 16..23 ; lanes 16-31: K 8..15 / 24..31.
static __device__ __forceinline__ void a_map(int lane, int h, int& ml, int& kl) {
  int hi = lane >> 4;
  ml = lane & 15;
  int v = h >> 1, lo = h & 1;
  kl = (v < 4) ? ((hi ? 8 : 0) + 2 * v + lo)
               : ((hi ? 24 : 16) + 2 * (v - 4) + lo);
}

// ---------------------------------------------------------------------------
// Kernel 1: pack W1 (as concatenated [256 x 512] PA|PB matrix), W2, F1 into
// f16 B-fragments (32x16 per fragment; lanes 0-15 hold K 0-7,16-23, lanes 16-31
// hold K 8-15,24-31; same K order as the A-matrix layout).
// grid = 512 blocks x 32 threads; one block packs one fragment.
// ---------------------------------------------------------------------------
__global__ void pack_weights_kernel(const float* __restrict__ W1,
                                    const float* __restrict__ W2,
                                    const float* __restrict__ F1,
                                    _Float16* __restrict__ w1p,
                                    _Float16* __restrict__ w2p,
                                    _Float16* __restrict__ f1p) {
  int f = blockIdx.x, lane = threadIdx.x;
  int nl = lane & 15;
  int kbase = (lane & 16) ? 16 : 0;
  if (f < 256) {                       // W1cat: K=256, N=512 -> 8 kt x 32 nt
    int kt = f >> 5, nt = f & 31;
    #pragma unroll
    for (int h = 0; h < 16; ++h) {
      int k = kt * 32 + ((h < 8) ? (kbase >> 1) + h : 16 + (kbase >> 1) + h - 8);
      int c = nt * 16 + nl;            // 0..511: cols 0-255 from W1 rows 0-255,
      float v = W1[(k + ((c >= 256) ? 256 : 0)) * 256 + (c & 255)];
      w1p[f * 512 + lane * 16 + h] = (_Float16)v;
    }
  } else {                             // W2 / F1: K=256, N=256 -> 8 kt x 16 nt
    int g = f - 256;
    const float* src = (g < 128) ? W2 : F1;
    _Float16* dst = (g < 128) ? w2p : f1p;
    int fl = g & 127;
    int kt = fl >> 4, nt = fl & 15;
    #pragma unroll
    for (int h = 0; h < 16; ++h) {
      int k = kt * 32 + ((h < 8) ? (kbase >> 1) + h : 16 + (kbase >> 1) + h - 8);
      float v = src[k * 256 + nt * 16 + nl];
      dst[fl * 512 + lane * 16 + h] = (_Float16)v;
    }
  }
}

// ---------------------------------------------------------------------------
// Kernel 2: projection GEMM. Virtual rows 0..5631 = 1536 shot rows then 4096
// query rows; out proj[5632][512] f32 (cols 0-255 = PA, 256-511 = PB).
// grid = 352 (row tiles of 16), block = 256 (8 waves); wave w does nt 4w..4w+3.
// ---------------------------------------------------------------------------
__global__ void proj_kernel(const float* __restrict__ x_shot,
                            const float* __restrict__ x_query,
                            const _Float16* __restrict__ w1p,
                            float* __restrict__ proj) {
  __shared__ float rows[16 * 256];
  __shared__ alignas(64) _Float16 afrag[8 * 512];
  int mt = blockIdx.x, t = threadIdx.x;
  {
    int row = t >> 4, c0 = (t & 15) << 4;
    int r = mt * 16 + row;
    const float* src = (r < 1536) ? (x_shot + r * 256) : (x_query + (r - 1536) * 256);
    #pragma unroll
    for (int c = 0; c < 16; ++c) rows[row * 256 + c0 + c] = src[c0 + c];
  }
  __syncthreads();
  {
    int frag = t >> 5, lane = t & 31;  // 8 fragments, one (frag,lane) per thread
    #pragma unroll
    for (int h = 0; h < 16; ++h) {
      int ml, kl; a_map(lane, h, ml, kl);
      afrag[frag * 512 + lane * 16 + h] = (_Float16)rows[ml * 256 + frag * 32 + kl];
    }
  }
  __syncthreads();
  int wave = t >> 5, lane = t & 31;
  int nloc = lane & 15, hi = lane >> 4;
  const v16h* A = (const v16h*)afrag;
  const v16h* Bp = (const v16h*)w1p;
  for (int s = 0; s < 4; ++s) {
    int nt = wave * 4 + s;
    v8f acc = {0.f, 0.f, 0.f, 0.f, 0.f, 0.f, 0.f, 0.f};
    #pragma unroll
    for (int kt = 0; kt < 8; ++kt) {
      v16h a = A[kt * 32 + lane];
      v16h b = Bp[(kt * 32 + nt) * 32 + lane];
      acc = WMMA_F16(a, b, acc);
    }
    int col = nt * 16 + nloc;
    #pragma unroll
    for (int r = 0; r < 8; ++r)
      proj[(mt * 16 + r + hi * 8) * 512 + col] = acc[r];
  }
}

// ---------------------------------------------------------------------------
// Kernel 3: shot-shot relations. grid = 256 groups (b,way); 30 pair rows
// (pad to 32), h1 = relu(PA[j]+PB[i]+b1), h2 = relu(h1@W2+b2), sum rows.
// Pad correction: subtract 2*relu(b2[col]).
// ---------------------------------------------------------------------------
__global__ void shot_rel_kernel(const float* __restrict__ proj,
                                const float* __restrict__ b1,
                                const float* __restrict__ b2,
                                const _Float16* __restrict__ w2p,
                                float* __restrict__ S_shot) {
  __shared__ float h1[32 * 256];
  __shared__ alignas(64) _Float16 afrag[16 * 512];
  int g = blockIdx.x, t = threadIdx.x;
  {
    float pa[6], pb[6];
    int base = g * 6;
    #pragma unroll
    for (int k = 0; k < 6; ++k) {
      pa[k] = proj[(base + k) * 512 + t];
      pb[k] = proj[(base + k) * 512 + 256 + t];
    }
    float bb = b1[t];
    #pragma unroll
    for (int p = 0; p < 30; ++p) {
      int i = p / 5, j = p % 5; j += (j >= i);   // ordered off-diag pairs, i-major
      float v = pa[j] + pb[i] + bb;
      h1[p * 256 + t] = v > 0.f ? v : 0.f;
    }
    h1[30 * 256 + t] = 0.f;
    h1[31 * 256 + t] = 0.f;
  }
  __syncthreads();
  for (int e = t; e < 16 * 32; e += 256) {
    int frag = e >> 5, lane = e & 31;
    int mt2 = frag >> 3, kt = frag & 7;
    #pragma unroll
    for (int h = 0; h < 16; ++h) {
      int ml, kl; a_map(lane, h, ml, kl);
      afrag[frag * 512 + lane * 16 + h] =
          (_Float16)h1[(mt2 * 16 + ml) * 256 + kt * 32 + kl];
    }
  }
  __syncthreads();
  int wave = t >> 5, lane = t & 31;
  int nloc = lane & 15, hi = lane >> 4;
  const v16h* A = (const v16h*)afrag;
  const v16h* Bp = (const v16h*)w2p;
  for (int u = 0; u < 2; ++u) {
    int nt = wave + u * 8;
    int col = nt * 16 + nloc;
    float bcol = b2[col];
    float ssum = 0.f;
    for (int mt2 = 0; mt2 < 2; ++mt2) {
      v8f acc = {0.f, 0.f, 0.f, 0.f, 0.f, 0.f, 0.f, 0.f};
      #pragma unroll
      for (int kt = 0; kt < 8; ++kt) {
        v16h a = A[(mt2 * 8 + kt) * 32 + lane];
        v16h b = Bp[(kt * 16 + nt) * 32 + lane];
        acc = WMMA_F16(a, b, acc);
      }
      #pragma unroll
      for (int r = 0; r < 8; ++r) {
        float h2 = acc[r] + bcol;
        ssum += h2 > 0.f ? h2 : 0.f;
      }
    }
    ssum += __shfl_xor(ssum, 16, 32);   // fold M-halves (lanes L and L+16 share col)
    if (hi == 0) {
      float pad = bcol > 0.f ? bcol : 0.f;
      S_shot[g * 256 + col] = ssum - 2.f * pad;
    }
  }
}

// ---------------------------------------------------------------------------
// Kernel 4 (dominant): query-involving relations. grid = 8192 groups
// (b,q,way); 12 pair rows padded to 16; adds hoisted S_shot; pad correction
// subtracts 4*relu(b2[col]).
// ---------------------------------------------------------------------------
__global__ void query_rel_kernel(const float* __restrict__ proj,
                                 const float* __restrict__ b1,
                                 const float* __restrict__ b2,
                                 const _Float16* __restrict__ w2p,
                                 const float* __restrict__ S_shot,
                                 float* __restrict__ rel_sum) {
  __shared__ float h1[16 * 256];
  __shared__ alignas(64) _Float16 afrag[8 * 512];
  int g = blockIdx.x, t = threadIdx.x;
  int b = g >> 6, rem = g & 63, q = rem >> 1, way = rem & 1;
  int g2 = b * 2 + way;
  {
    int sbase = g2 * 6;
    int qrow = 1536 + b * 32 + q;
    float paq = proj[qrow * 512 + t];
    float pbq = proj[qrow * 512 + 256 + t];
    float bb = b1[t];
    #pragma unroll
    for (int k = 0; k < 6; ++k) {
      float pas = proj[(sbase + k) * 512 + t];
      float pbs = proj[(sbase + k) * 512 + 256 + t];
      float v0 = paq + pbs + bb;   // pair (i=shot k, j=query)
      float v1 = pas + pbq + bb;   // pair (i=query, j=shot k)
      h1[k * 256 + t] = v0 > 0.f ? v0 : 0.f;
      h1[(6 + k) * 256 + t] = v1 > 0.f ? v1 : 0.f;
    }
    #pragma unroll
    for (int p = 12; p < 16; ++p) h1[p * 256 + t] = 0.f;
  }
  __syncthreads();
  {
    int frag = t >> 5, lane = t & 31;
    #pragma unroll
    for (int h = 0; h < 16; ++h) {
      int ml, kl; a_map(lane, h, ml, kl);
      afrag[frag * 512 + lane * 16 + h] =
          (_Float16)h1[ml * 256 + frag * 32 + kl];
    }
  }
  __syncthreads();
  int wave = t >> 5, lane = t & 31;
  int nloc = lane & 15, hi = lane >> 4;
  const v16h* A = (const v16h*)afrag;
  const v16h* Bp = (const v16h*)w2p;
  for (int u = 0; u < 2; ++u) {
    int nt = wave + u * 8;
    int col = nt * 16 + nloc;
    float bcol = b2[col];
    v8f acc = {0.f, 0.f, 0.f, 0.f, 0.f, 0.f, 0.f, 0.f};
    #pragma unroll
    for (int kt = 0; kt < 8; ++kt) {
      v16h a = A[kt * 32 + lane];
      v16h b = Bp[(kt * 16 + nt) * 32 + lane];
      acc = WMMA_F16(a, b, acc);
    }
    float ssum = 0.f;
    #pragma unroll
    for (int r = 0; r < 8; ++r) {
      float h2 = acc[r] + bcol;
      ssum += h2 > 0.f ? h2 : 0.f;
    }
    ssum += __shfl_xor(ssum, 16, 32);
    if (hi == 0) {
      float pad = bcol > 0.f ? bcol : 0.f;
      rel_sum[g * 256 + col] = ssum - 4.f * pad + S_shot[g2 * 256 + col];
    }
  }
}

// ---------------------------------------------------------------------------
// Kernel 5: head. f = relu(rel_sum@F1+fb1); logit = f.F2 + fb2.
// grid = 512 row tiles of 16; GEMM fused with the F2 dot via shuffle-tree +
// per-wave LDS partials (deterministic, no float atomics).
// ---------------------------------------------------------------------------
__global__ void head_kernel(const float* __restrict__ rel_sum,
                            const _Float16* __restrict__ f1p,
                            const float* __restrict__ fb1,
                            const float* __restrict__ F2,
                            const float* __restrict__ fb2,
                            float* __restrict__ out) {
  __shared__ float rows[16 * 256];
  __shared__ alignas(64) _Float16 afrag[8 * 512];
  __shared__ float wred[8 * 16];
  int mt = blockIdx.x, t = threadIdx.x;
  {
    int row = t >> 4, c0 = (t & 15) << 4;
    #pragma unroll
    for (int c = 0; c < 16; ++c)
      rows[row * 256 + c0 + c] = rel_sum[(mt * 16 + row) * 256 + c0 + c];
  }
  __syncthreads();
  {
    int frag = t >> 5, lane = t & 31;
    #pragma unroll
    for (int h = 0; h < 16; ++h) {
      int ml, kl; a_map(lane, h, ml, kl);
      afrag[frag * 512 + lane * 16 + h] =
          (_Float16)rows[ml * 256 + frag * 32 + kl];
    }
  }
  __syncthreads();
  int wave = t >> 5, lane = t & 31;
  int nloc = lane & 15, hi = lane >> 4;
  const v16h* A = (const v16h*)afrag;
  const v16h* Bp = (const v16h*)f1p;
  float pr[8];
  #pragma unroll
  for (int r = 0; r < 8; ++r) pr[r] = 0.f;
  for (int u = 0; u < 2; ++u) {
    int nt = wave + u * 8;
    int col = nt * 16 + nloc;
    v8f acc = {0.f, 0.f, 0.f, 0.f, 0.f, 0.f, 0.f, 0.f};
    #pragma unroll
    for (int kt = 0; kt < 8; ++kt) {
      v16h a = A[kt * 32 + lane];
      v16h b = Bp[(kt * 16 + nt) * 32 + lane];
      acc = WMMA_F16(a, b, acc);
    }
    float fb = fb1[col], w = F2[col];
    #pragma unroll
    for (int r = 0; r < 8; ++r) {
      float f = acc[r] + fb;
      pr[r] += (f > 0.f ? f : 0.f) * w;
    }
  }
  #pragma unroll
  for (int m = 1; m < 16; m <<= 1) {
    #pragma unroll
    for (int r = 0; r < 8; ++r) pr[r] += __shfl_xor(pr[r], m, 32);
  }
  if (nloc == 0) {
    #pragma unroll
    for (int r = 0; r < 8; ++r) wred[wave * 16 + hi * 8 + r] = pr[r];
  }
  __syncthreads();
  if (t < 16) {
    float s = 0.f;
    #pragma unroll
    for (int w = 0; w < 8; ++w) s += wred[w * 16 + t];
    out[mt * 16 + t] = s + fb2[0];   // out row = b*64 + q*2 + way  ([B,Q,1,2])
  }
}

// ---------------------------------------------------------------------------
// Workspace layout (bytes, ~19.8 MB total):
//   [0,       262144)  W1 packed fragments (f16, 256 frags x 1KB)
//   [262144,  393216)  W2 packed
//   [393216,  524288)  F1 packed
//   [524288, 12058624) proj [5632][512] f32
//   [12058624,12320768) S_shot [256][256] f32
//   [12320768,20709376) rel_sum [8192][256] f32
// ---------------------------------------------------------------------------
extern "C" void kernel_launch(void* const* d_in, const int* in_sizes, int n_in,
                              void* d_out, int out_size, void* d_ws, size_t ws_size,
                              hipStream_t stream) {
  const float* x_shot  = (const float*)d_in[0];
  const float* x_query = (const float*)d_in[1];
  const float* W1  = (const float*)d_in[2];
  const float* b1  = (const float*)d_in[3];
  const float* W2  = (const float*)d_in[4];
  const float* b2  = (const float*)d_in[5];
  const float* F1  = (const float*)d_in[6];
  const float* fb1 = (const float*)d_in[7];
  const float* F2  = (const float*)d_in[8];
  const float* fb2 = (const float*)d_in[9];

  char* ws = (char*)d_ws;
  _Float16* w1p   = (_Float16*)(ws + 0);
  _Float16* w2p   = (_Float16*)(ws + 262144);
  _Float16* f1p   = (_Float16*)(ws + 393216);
  float* proj     = (float*)(ws + 524288);
  float* S_shot   = (float*)(ws + 12058624);
  float* rel_sum  = (float*)(ws + 12320768);
  float* out = (float*)d_out;

  pack_weights_kernel<<<512, 32, 0, stream>>>(W1, W2, F1, w1p, w2p, f1p);
  proj_kernel<<<352, 256, 0, stream>>>(x_shot, x_query, w1p, proj);
  shot_rel_kernel<<<256, 256, 0, stream>>>(proj, b1, b2, w2p, S_shot);
  query_rel_kernel<<<8192, 256, 0, stream>>>(proj, b1, b2, w2p, S_shot, rel_sum);
  head_kernel<<<512, 256, 0, stream>>>(rel_sum, f1p, fb1, F2, fb2, out);

  (void)in_sizes; (void)n_in; (void)out_size; (void)ws_size;
}
